// GroupQueryAttention_12481174962867
// MI455X (gfx1250) — matrix operand order repaired and checked by reading, not compiled, not run on hardware
//
#include <hip/hip_runtime.h>
#include <hip/hip_bf16.h>

typedef __attribute__((ext_vector_type(16))) __bf16 v16bf;
typedef __attribute__((ext_vector_type(8)))  __bf16 v8bf;
typedef __attribute__((ext_vector_type(8)))  float  v8f;
typedef int v4i_vs __attribute__((__vector_size__(16)));   // matches builtin param

#define D_   2048
#define QH_  32
#define KVH_ 8
#define HD_  64
#define B_   2
#define S_   2048
#define SCALE_ 0.125f                     // 1/sqrt(64)
#define LOG2_THETA 19.931568569324174f    // log2(1e6)

// ---- gfx1250 async global->LDS copies (ASYNCcnt path), guarded -----------
#if __has_builtin(__builtin_amdgcn_global_load_async_to_lds_b128)
#define USE_ASYNC_LDS 1
#else
#define USE_ASYNC_LDS 0
#endif

template<int OFF>
static __device__ __forceinline__ void async_copy16(void* l, const void* g) {
#if USE_ASYNC_LDS
  __builtin_amdgcn_global_load_async_to_lds_b128(
      (__attribute__((address_space(1))) v4i_vs*)g,
      (__attribute__((address_space(3))) v4i_vs*)l, OFF, 0);
#else
  *(uint4*)((char*)l + OFF) = *(const uint4*)((const char*)g + OFF);
#endif
}

static __device__ __forceinline__ void async_wait0() {
#if USE_ASYNC_LDS
#if __has_builtin(__builtin_amdgcn_s_wait_asynccnt)
  __builtin_amdgcn_s_wait_asynccnt(0);
#else
  asm volatile("s_wait_asynccnt 0x0" ::: "memory");
#endif
#endif
}

static __device__ __forceinline__ v16bf cat8(v8bf lo, v8bf hi) {
  return __builtin_shufflevector(lo, hi, 0,1,2,3,4,5,6,7,8,9,10,11,12,13,14,15);
}

// ---------------------------------------------------------------------------
// Weight convert + transpose: W[k][n] fp32 -> Wt[n][k] bf16
// ---------------------------------------------------------------------------
__global__ void wt_convert_transpose(const float* __restrict__ W,
                                     __bf16* __restrict__ Wt, int K, int N) {
  long idx = (long)blockIdx.x * 256 + threadIdx.x;
  long total = (long)K * N;
  if (idx >= total) return;
  int n = (int)(idx / K);
  int k = (int)(idx % K);
  Wt[idx] = (__bf16)W[(long)k * N + n];
}

// ---------------------------------------------------------------------------
// WMMA GEMM: C(MxN) = A(MxK) @ W(KxN), W given transposed bf16 Wt[n][k].
// Block: 256 threads = 8 waves; tile 128x128; wave tile 32(M) x 64(N).
// MODE 1: RoPE epilogue -> Qh[b][h][s][hd]       (nheads = QH)
// MODE 2: RoPE epilogue -> Kh[b][kvh][s][hd]     (nheads = KVH)
// MODE 3: transpose     -> Vt[b][kvh][hd][s]     (nheads = KVH)
// MODE 4: fp32 + bias   -> out[r][c]
// ---------------------------------------------------------------------------
template<int MODE, bool ABF>
__global__ __launch_bounds__(256, 1)
void gemm_wmma(const void* __restrict__ Ap,
               const __bf16* __restrict__ Bt,
               void* __restrict__ Cp,
               const float* __restrict__ bias,
               int M, int N, int K, int nheads)
{
  __shared__ __align__(16) __bf16 lA[128][40];   // 80B row stride (16B multiple)
  __shared__ __align__(16) __bf16 lB[128][40];

  const int t    = threadIdx.x;
  const int lane = t & 31;
  const int wave = t >> 5;
  const int wy   = wave & 3;     // 0..3 along M
  const int wx   = wave >> 2;    // 0..1 along N
  const int hk   = lane >> 4;    // half-wave select
  const int ln   = lane & 15;
  const int m0   = blockIdx.y * 128;
  const int n0   = blockIdx.x * 128;

  v8f acc[2][4];
  #pragma unroll
  for (int m = 0; m < 2; ++m)
    #pragma unroll
    for (int j = 0; j < 4; ++j)
      acc[m][j] = (v8f){};

  const int rowL = t >> 1;          // 0..127
  const int segL = (t & 1) * 16;    // 0 or 16

  for (int k0 = 0; k0 < K; k0 += 32) {
    // ---- stage A tile (128 x 32) ----
    if constexpr (ABF) {
      const __bf16* src = (const __bf16*)Ap + (long)(m0 + rowL) * K + k0 + segL;
      async_copy16<0> (&lA[rowL][segL], src);
      async_copy16<16>(&lA[rowL][segL], src);
    } else {
      const float* src = (const float*)Ap + (long)(m0 + rowL) * K + k0 + segL;
      float4 f0 = ((const float4*)src)[0];
      float4 f1 = ((const float4*)src)[1];
      float4 f2 = ((const float4*)src)[2];
      float4 f3 = ((const float4*)src)[3];
      v8bf h0 = { (__bf16)f0.x,(__bf16)f0.y,(__bf16)f0.z,(__bf16)f0.w,
                  (__bf16)f1.x,(__bf16)f1.y,(__bf16)f1.z,(__bf16)f1.w };
      v8bf h1 = { (__bf16)f2.x,(__bf16)f2.y,(__bf16)f2.z,(__bf16)f2.w,
                  (__bf16)f3.x,(__bf16)f3.y,(__bf16)f3.z,(__bf16)f3.w };
      *(v8bf*)&lA[rowL][segL]     = h0;
      *(v8bf*)&lA[rowL][segL + 8] = h1;
    }
    // ---- stage B tile (128 x 32), Wt rows contiguous in K ----
    {
      const __bf16* src = Bt + (long)(n0 + rowL) * K + k0 + segL;
      async_copy16<0> (&lB[rowL][segL], src);
      async_copy16<16>(&lB[rowL][segL], src);
    }
    async_wait0();
    __syncthreads();

    // ---- fragments + 8 WMMA ----
    v16bf afr[2], bfr[4];
    #pragma unroll
    for (int m = 0; m < 2; ++m) {
      const __bf16* p = &lA[wy*32 + m*16 + ln][hk * 8];
      afr[m] = cat8(*(const v8bf*)p, *(const v8bf*)(p + 16));
    }
    #pragma unroll
    for (int j = 0; j < 4; ++j) {
      const __bf16* p = &lB[wx*64 + j*16 + ln][hk * 8];
      bfr[j] = cat8(*(const v8bf*)p, *(const v8bf*)(p + 16));
    }
    #pragma unroll
    for (int m = 0; m < 2; ++m)
      #pragma unroll
      for (int j = 0; j < 4; ++j)
        acc[m][j] = __builtin_amdgcn_wmma_f32_16x16x32_bf16(
            false, afr[m], false, bfr[j], (short)0, acc[m][j], false, false);
    __syncthreads();
  }

  // ---- epilogues ----
  #pragma unroll
  for (int m = 0; m < 2; ++m) {
    const int rbase = m0 + wy*32 + m*16 + hk*8;
    if constexpr (MODE == 4) {
      float* out = (float*)Cp;
      #pragma unroll
      for (int j = 0; j < 4; ++j) {
        const int c = n0 + wx*64 + j*16 + ln;
        const float bv = bias[c];
        #pragma unroll
        for (int i = 0; i < 8; ++i)
          out[(long)(rbase + i) * N + c] = acc[m][j][i] + bv;
      }
    } else if constexpr (MODE == 3) {
      __bf16* Vt = (__bf16*)Cp;
      const int h = (n0 >> 6) + wx;           // head within KVH
      #pragma unroll
      for (int j = 0; j < 4; ++j) {
        const int hd = j*16 + ln;
        #pragma unroll
        for (int i = 0; i < 8; ++i) {
          const int r  = rbase + i;
          const int bb = r >> 11;             // r / S
          const int s  = r & (S_ - 1);
          Vt[((long)((bb*KVH_ + h)*HD_ + hd))*S_ + s] = (__bf16)acc[m][j][i];
        }
      }
    } else {  // MODE 1 / 2: RoPE, pairs (hd, hd+32) live in accum tiles (j, j+2)
      __bf16* Out = (__bf16*)Cp;
      const int h = (n0 >> 6) + wx;
      #pragma unroll
      for (int j = 0; j < 2; ++j) {
        const int hd = j*16 + ln;             // 0..31
        const float invf = exp2f(-(float)hd * (LOG2_THETA / 32.0f));
        #pragma unroll
        for (int i = 0; i < 8; ++i) {
          const int r  = rbase + i;
          const int bb = r >> 11;
          const int s  = r & (S_ - 1);
          const float ang = (float)s * invf;
          const float cs = cosf(ang), sn = sinf(ang);
          const float x1 = acc[m][j][i], x2 = acc[m][j+2][i];
          const long base = ((long)(bb*nheads + h)*S_ + s) * HD_;
          Out[base + hd]      = (__bf16)(x1*cs - x2*sn);
          Out[base + hd + 32] = (__bf16)(x2*cs + x1*sn);
        }
      }
    }
  }
}

// ---------------------------------------------------------------------------
// Flash attention: grid (S/128, QH, B), 256 threads = 8 waves x 16 q-rows.
// Qh[b][h][s][hd], Kh[b][kvh][s][hd], Vt[b][kvh][hd][s] -> Oh[b][s][h*64+hd]
// ---------------------------------------------------------------------------
__global__ __launch_bounds__(256, 1)
void flash_attn(const __bf16* __restrict__ Qh,
                const __bf16* __restrict__ Kh,
                const __bf16* __restrict__ Vt,
                __bf16* __restrict__ Oh)
{
  __shared__ __align__(16) __bf16 lK[64][64];       // [key][hd]
  __shared__ __align__(16) __bf16 lV[64][64];       // [hd][key]
  __shared__ __align__(16) __bf16 lP[8][16][64];    // per-wave P tile [row][key]

  const int t    = threadIdx.x;
  const int lane = t & 31;
  const int wave = t >> 5;                 // 0..7
  const int hk   = lane >> 4;
  const int ln   = lane & 15;
  const int b    = blockIdx.z;
  const int h    = blockIdx.y;
  const int kvh  = h >> 2;                 // QH/KVH = 4
  const int q0   = blockIdx.x * 128 + wave * 16;

  // Q fragments held in registers (2 k-steps over HD)
  v16bf aQ[2];
  #pragma unroll
  for (int ks = 0; ks < 2; ++ks) {
    const __bf16* p = Qh + ((long)((b*QH_ + h)*S_) + q0 + ln) * HD_ + ks*32 + hk*8;
    aQ[ks] = cat8(*(const v8bf*)p, *(const v8bf*)(p + 16));
  }

  float rm[8], rs[8];
  v8f accO[4];
  #pragma unroll
  for (int i = 0; i < 8; ++i) { rm[i] = -1e30f; rs[i] = 0.f; }
  #pragma unroll
  for (int jo = 0; jo < 4; ++jo) accO[jo] = (v8f){};

  const int crow = t >> 2;            // 0..63
  const int cseg = (t & 3) * 16;      // 0,16,32,48
  const __bf16* Ksrc0 = Kh + (long)((b*KVH_ + kvh)*S_) * HD_;
  const __bf16* Vsrc0 = Vt + (long)((b*KVH_ + kvh)*HD_) * S_;

  for (int kt = 0; kt < S_ / 64; ++kt) {
    // ---- prefetch next tile (gfx1250 global_prefetch_b8) ----
    if (kt + 1 < S_ / 64) {
      __builtin_prefetch(Ksrc0 + (long)((kt + 1)*64 + crow) * HD_ + cseg, 0, 1);
      __builtin_prefetch(Vsrc0 + (long)crow * S_ + (kt + 1)*64 + cseg, 0, 1);
    }
    // ---- stage K/V 64x64 tiles (32B per thread each, async to LDS) ----
    {
      const __bf16* src = Ksrc0 + (long)(kt*64 + crow) * HD_ + cseg;
      async_copy16<0> (&lK[crow][cseg], src);
      async_copy16<16>(&lK[crow][cseg], src);
    }
    {
      const __bf16* src = Vsrc0 + (long)crow * S_ + kt*64 + cseg;
      async_copy16<0> (&lV[crow][cseg], src);
      async_copy16<16>(&lV[crow][cseg], src);
    }
    async_wait0();
    __syncthreads();

    // ---- scores S = Q K^T (16 x 64) ----
    v8f sc[4];
    #pragma unroll
    for (int j = 0; j < 4; ++j) sc[j] = (v8f){};
    #pragma unroll
    for (int ks = 0; ks < 2; ++ks)
      #pragma unroll
      for (int j = 0; j < 4; ++j) {
        const __bf16* p = &lK[j*16 + ln][ks*32 + hk*8];
        v16bf bK = cat8(*(const v8bf*)p, *(const v8bf*)(p + 16));
        sc[j] = __builtin_amdgcn_wmma_f32_16x16x32_bf16(
            false, aQ[ks], false, bK, (short)0, sc[j], false, false);
      }

    // ---- online softmax update ----
    #pragma unroll
    for (int i = 0; i < 8; ++i) {
      float lm = -1e30f;
      #pragma unroll
      for (int j = 0; j < 4; ++j) lm = fmaxf(lm, sc[j][i]);
      lm *= SCALE_;
      lm = fmaxf(lm, __shfl_xor(lm, 1, 32));
      lm = fmaxf(lm, __shfl_xor(lm, 2, 32));
      lm = fmaxf(lm, __shfl_xor(lm, 4, 32));
      lm = fmaxf(lm, __shfl_xor(lm, 8, 32));
      const float mo = rm[i];
      const float mn = fmaxf(mo, lm);
      const float alpha = __expf(mo - mn);
      float ps = 0.f;
      #pragma unroll
      for (int j = 0; j < 4; ++j) {
        const float p = __expf(sc[j][i] * SCALE_ - mn);
        sc[j][i] = p;
        ps += p;
      }
      ps += __shfl_xor(ps, 1, 32);
      ps += __shfl_xor(ps, 2, 32);
      ps += __shfl_xor(ps, 4, 32);
      ps += __shfl_xor(ps, 8, 32);
      rs[i] = rs[i] * alpha + ps;
      rm[i] = mn;
      #pragma unroll
      for (int jo = 0; jo < 4; ++jo) accO[jo][i] *= alpha;
    }

    // ---- P: C-layout -> A-layout via LDS ----
    #pragma unroll
    for (int j = 0; j < 4; ++j)
      #pragma unroll
      for (int i = 0; i < 8; ++i)
        lP[wave][hk*8 + i][j*16 + ln] = (__bf16)sc[j][i];
    __syncthreads();

    // ---- O += P V ----
    #pragma unroll
    for (int ks2 = 0; ks2 < 2; ++ks2) {
      const __bf16* pa = &lP[wave][ln][ks2*32 + hk*8];
      v16bf aP = cat8(*(const v8bf*)pa, *(const v8bf*)(pa + 16));
      #pragma unroll
      for (int jo = 0; jo < 4; ++jo) {
        const __bf16* p = &lV[jo*16 + ln][ks2*32 + hk*8];
        v16bf bV = cat8(*(const v8bf*)p, *(const v8bf*)(p + 16));
        accO[jo] = __builtin_amdgcn_wmma_f32_16x16x32_bf16(
            false, aP, false, bV, (short)0, accO[jo], false, false);
      }
    }
    __syncthreads();
  }

  // ---- normalize + write Oh[b][s][h*64+hd] ----
  #pragma unroll
  for (int jo = 0; jo < 4; ++jo)
    #pragma unroll
    for (int i = 0; i < 8; ++i) {
      const int s  = q0 + hk*8 + i;
      const int hd = jo*16 + ln;
      Oh[((long)(b*S_ + s))*D_ + h*HD_ + hd] = (__bf16)(accO[jo][i] / rs[i]);
    }
}

// ---------------------------------------------------------------------------
extern "C" void kernel_launch(void* const* d_in, const int* in_sizes, int n_in,
                              void* d_out, int out_size, void* d_ws, size_t ws_size,
                              hipStream_t stream)
{
  (void)in_sizes; (void)n_in; (void)out_size; (void)ws_size;
  const float* q  = (const float*)d_in[0];
  const float* k  = (const float*)d_in[1];
  const float* v  = (const float*)d_in[2];
  // d_in[3] = attn_mask: unused by the reference computation.
  const float* Wq = (const float*)d_in[4];
  const float* Wk = (const float*)d_in[5];
  const float* Wv = (const float*)d_in[6];
  const float* Wo = (const float*)d_in[7];
  const float* bo = (const float*)d_in[8];

  char* ws = (char*)d_ws;
  size_t off = 0;
  auto alloc = [&](size_t bytes) -> char* {
    char* p = ws + off;
    off += (bytes + 255) & ~(size_t)255;
    return p;
  };
  __bf16* WqT = (__bf16*)alloc((size_t)D_ * D_ * 2);
  __bf16* WkT = (__bf16*)alloc((size_t)D_ * KVH_ * HD_ * 2);
  __bf16* WvT = (__bf16*)alloc((size_t)D_ * KVH_ * HD_ * 2);
  __bf16* WoT = (__bf16*)alloc((size_t)D_ * D_ * 2);
  __bf16* Qhb = (__bf16*)alloc((size_t)B_ * QH_  * S_ * HD_ * 2);
  __bf16* Khb = (__bf16*)alloc((size_t)B_ * KVH_ * S_ * HD_ * 2);
  __bf16* Vtb = (__bf16*)alloc((size_t)B_ * KVH_ * S_ * HD_ * 2);
  __bf16* Ohb = (__bf16*)alloc((size_t)B_ * S_ * D_ * 2);

  const int NKV = KVH_ * HD_;   // 512
  wt_convert_transpose<<<(D_*D_ + 255)/256, 256, 0, stream>>>(Wq, WqT, D_, D_);
  wt_convert_transpose<<<(D_*NKV + 255)/256, 256, 0, stream>>>(Wk, WkT, D_, NKV);
  wt_convert_transpose<<<(D_*NKV + 255)/256, 256, 0, stream>>>(Wv, WvT, D_, NKV);
  wt_convert_transpose<<<(D_*D_ + 255)/256, 256, 0, stream>>>(Wo, WoT, D_, D_);

  dim3 blk(256);
  const int MB = (B_ * S_) / 128;  // 32
  gemm_wmma<1,false><<<dim3(D_/128, MB), blk, 0, stream>>>(q, WqT, Qhb, nullptr,
                                                           B_*S_, D_, D_, QH_);
  gemm_wmma<2,false><<<dim3(NKV/128, MB), blk, 0, stream>>>(k, WkT, Khb, nullptr,
                                                            B_*S_, NKV, D_, KVH_);
  gemm_wmma<3,false><<<dim3(NKV/128, MB), blk, 0, stream>>>(v, WvT, Vtb, nullptr,
                                                            B_*S_, NKV, D_, KVH_);
  flash_attn<<<dim3(S_/128, QH_, B_), dim3(256), 0, stream>>>(Qhb, Khb, Vtb, Ohb);
  gemm_wmma<4,true><<<dim3(D_/128, MB), blk, 0, stream>>>(Ohb, WoT, d_out, bo,
                                                          B_*S_, D_, D_, 0);
}
